// PairFormerStack_2963527434489
// MI455X (gfx1250) — matrix operand order repaired and checked by reading, not compiled, not run on hardware
//
#include <hip/hip_runtime.h>

// ---------------------------------------------------------------------------
// PairFormer stack for MI455X (gfx1250, wave32, WMMA bf16 16x16x32, f32 acc)
// LDS operand tiles are stored in WMMA *fragment order* so every fragment is
// two ds_load_b128 per wave instead of 32 scalar ds_load_u16 gathers.
//
// Fragment layouts (wave32):
//   A (16x32 bf16): lane = (m&15) + 16*((k>>3)&1), elem e = (k&7) + (k>=16?8:0)
//   B (32x16 bf16): lane = (n&15) + 16*((k>>3)&1), elem e = (k&7) + (k>=16?8:0)
//   C/D (16x16 f32): n = lane&15, m = v + 8*(lane>>4)
// ---------------------------------------------------------------------------

typedef __attribute__((ext_vector_type(16))) __bf16 v16bf;
typedef __attribute__((ext_vector_type(8)))  __bf16 v8bf;
typedef __attribute__((ext_vector_type(8)))  float  v8f;

#define WMMA_BF16(a, b, c) \
  __builtin_amdgcn_wmma_f32_16x16x32_bf16(false, (a), false, (b), (short)0, (c), false, false)

#define NTOK 256
#define CZd 128
static const size_t PAIR_ELEMS = (size_t)NTOK * NTOK * CZd;  // 8388608

__device__ __forceinline__ float sig_(float x) { return 1.0f / (1.0f + __expf(-x)); }

// ---------------------------- LayerNorm (rows) -----------------------------
__global__ __launch_bounds__(128) void k_ln(const float* __restrict__ x,
                                            const float* __restrict__ g,
                                            const float* __restrict__ b,
                                            float* __restrict__ y, int C) {
  __shared__ float sred[128];
  const int row = blockIdx.x, tid = threadIdx.x;
  const float4* xr = (const float4*)(x + (size_t)row * C);
  float4* yr = (float4*)(y + (size_t)row * C);
  const int C4 = C >> 2;
  float acc = 0.f;
  for (int c = tid; c < C4; c += 128) {
    float4 v = xr[c];
    acc += v.x + v.y + v.z + v.w;
  }
  sred[tid] = acc; __syncthreads();
  for (int o = 64; o > 0; o >>= 1) { if (tid < o) sred[tid] += sred[tid + o]; __syncthreads(); }
  const float mean = sred[0] / C;
  __syncthreads();
  float var = 0.f;
  for (int c = tid; c < C4; c += 128) {
    float4 v = xr[c];
    float dx = v.x - mean, dy = v.y - mean, dz = v.z - mean, dw = v.w - mean;
    var += dx * dx + dy * dy + dz * dz + dw * dw;
  }
  sred[tid] = var; __syncthreads();
  for (int o = 64; o > 0; o >>= 1) { if (tid < o) sred[tid] += sred[tid + o]; __syncthreads(); }
  const float inv = rsqrtf(sred[0] / C + 1e-5f);
  const float4* g4 = (const float4*)g;
  const float4* b4 = (const float4*)b;
  for (int c = tid; c < C4; c += 128) {
    float4 v = xr[c], gv = g4[c], bv = b4[c], o4;
    o4.x = (v.x - mean) * inv * gv.x + bv.x;
    o4.y = (v.y - mean) * inv * gv.y + bv.y;
    o4.z = (v.z - mean) * inv * gv.z + bv.z;
    o4.w = (v.w - mean) * inv * gv.w + bv.w;
    yr[c] = o4;
  }
}

// --------------------- generic fused WMMA GEMM (f32 I/O) -------------------
// P1 = A1@W1 (+colbias), P2 = A2@W2
// act: 0 -> P1 ; 1 -> sigmoid(P1)*P2 ; 2 -> silu(P1)*P2 ; 3 -> sigmoid(P1)
// then *rowscale[m] (opt), +residual[m,n] (opt). M%64==0, K%32==0; N guarded.
__global__ __launch_bounds__(256) void k_gemm(
    const float* __restrict__ A1, const float* __restrict__ W1,
    const float* __restrict__ A2, const float* __restrict__ W2,
    const float* __restrict__ colbias, const float* __restrict__ rowscale,
    const float* residual, float* Y, int M, int N, int K, int act) {
  __shared__ __align__(32) __bf16 As1[4 * 512];  // [mi][lane][e]
  __shared__ __align__(32) __bf16 Ws1[4 * 512];  // [nt][lane][e]
  __shared__ __align__(32) __bf16 As2[4 * 512];
  __shared__ __align__(32) __bf16 Ws2[4 * 512];

  const int tid = threadIdx.x, lane = tid & 31, wave = tid >> 5;
  const int half = lane >> 4, nl = lane & 15;
  const int m0 = blockIdx.y * 64, n0 = blockIdx.x * 64;
  const bool dual = (act == 1 || act == 2);
  const int mi = wave >> 1;            // 0..3
  const int njb = (wave & 1) * 2;      // 0 or 2

  // A staging map: thread -> row r = tid>>2, col group cg = tid&3 (8 consecutive k)
  const int a_r = tid >> 2, a_cg = tid & 3;
  const int a_dst = (a_r >> 4) * 512 + ((a_r & 15) + 16 * (a_cg & 1)) * 16 + ((a_cg >= 2) ? 8 : 0);
  // W staging map: thread -> col n = tid&63, k group kg = tid>>6 (8 consecutive k)
  const int w_n = tid & 63, w_kg = tid >> 6;
  const int w_dst = (w_n >> 4) * 512 + ((w_n & 15) + 16 * (w_kg & 1)) * 16 + (w_kg >> 1) * 8;

  v8f acc1a = {}, acc1b = {}, acc2a = {}, acc2b = {};

  for (int k0 = 0; k0 < K; k0 += 32) {
    {
      const float* src = A1 + (size_t)(m0 + a_r) * K + k0 + a_cg * 8;
      float4 f0 = *(const float4*)src, f1 = *(const float4*)(src + 4);
      v8bf t;
      t[0] = (__bf16)f0.x; t[1] = (__bf16)f0.y; t[2] = (__bf16)f0.z; t[3] = (__bf16)f0.w;
      t[4] = (__bf16)f1.x; t[5] = (__bf16)f1.y; t[6] = (__bf16)f1.z; t[7] = (__bf16)f1.w;
      *(v8bf*)(As1 + a_dst) = t;
      if (dual) {
        const float* s2 = A2 + (size_t)(m0 + a_r) * K + k0 + a_cg * 8;
        float4 g0 = *(const float4*)s2, g1 = *(const float4*)(s2 + 4);
        v8bf u;
        u[0] = (__bf16)g0.x; u[1] = (__bf16)g0.y; u[2] = (__bf16)g0.z; u[3] = (__bf16)g0.w;
        u[4] = (__bf16)g1.x; u[5] = (__bf16)g1.y; u[6] = (__bf16)g1.z; u[7] = (__bf16)g1.w;
        *(v8bf*)(As2 + a_dst) = u;
      }
    }
    {
      const int gc = n0 + w_n;
      v8bf t, u;
      for (int i = 0; i < 8; ++i) {
        const int kk = k0 + w_kg * 8 + i;
        t[i] = (__bf16)((gc < N) ? W1[(size_t)kk * N + gc] : 0.0f);
        if (dual) u[i] = (__bf16)((gc < N) ? W2[(size_t)kk * N + gc] : 0.0f);
      }
      *(v8bf*)(Ws1 + w_dst) = t;
      if (dual) *(v8bf*)(Ws2 + w_dst) = u;
    }
    __syncthreads();

    const int lo = lane * 16;
    {
      v16bf a = *(const v16bf*)(As1 + mi * 512 + lo);
      v16bf ba = *(const v16bf*)(Ws1 + njb * 512 + lo);
      v16bf bb = *(const v16bf*)(Ws1 + (njb + 1) * 512 + lo);
      acc1a = WMMA_BF16(a, ba, acc1a);
      acc1b = WMMA_BF16(a, bb, acc1b);
    }
    if (dual) {
      v16bf a = *(const v16bf*)(As2 + mi * 512 + lo);
      v16bf ba = *(const v16bf*)(Ws2 + njb * 512 + lo);
      v16bf bb = *(const v16bf*)(Ws2 + (njb + 1) * 512 + lo);
      acc2a = WMMA_BF16(a, ba, acc2a);
      acc2b = WMMA_BF16(a, bb, acc2b);
    }
    __syncthreads();
  }

  for (int t = 0; t < 2; ++t) {
    v8f p1 = t ? acc1b : acc1a;
    v8f p2 = t ? acc2b : acc2a;
    int gn = n0 + (njb + t) * 16 + nl;
    if (gn >= N) continue;
    for (int v = 0; v < 8; ++v) {
      int gm = m0 + mi * 16 + v + 8 * half;
      float x1 = p1[v];
      if (colbias) x1 += colbias[gn];
      float out;
      if (act == 0) out = x1;
      else if (act == 1) out = sig_(x1) * p2[v];
      else if (act == 2) out = x1 * sig_(x1) * p2[v];
      else out = sig_(x1);
      if (rowscale) out *= rowscale[gm];
      size_t oi = (size_t)gm * N + gn;
      if (residual) out += residual[oi];
      Y[oi] = out;
    }
  }
}

// ------- batched per-channel GEMM for triangle contraction (bf16 in) -------
// A packed [c][i][k], B packed [c][j][k]; Y[i,j,c] = sum_k A[i,k]*B[j,k]
__global__ __launch_bounds__(256) void k_tri_gemm(const __bf16* __restrict__ At,
                                                  const __bf16* __restrict__ Bt,
                                                  float* __restrict__ Y) {
  __shared__ __align__(32) __bf16 As[4 * 512];
  __shared__ __align__(32) __bf16 Bs[4 * 512];
  const int tid = threadIdx.x, lane = tid & 31, wave = tid >> 5;
  const int half = lane >> 4, nl = lane & 15;
  const int c = blockIdx.z;
  const int m0 = blockIdx.y * 64, n0 = blockIdx.x * 64;
  const __bf16* A = At + (size_t)c * NTOK * NTOK;
  const __bf16* B = Bt + (size_t)c * NTOK * NTOK;
  const int mi = wave >> 1, njb = (wave & 1) * 2;

  const int a_r = tid >> 2, a_cg = tid & 3;
  const int a_dst = (a_r >> 4) * 512 + ((a_r & 15) + 16 * (a_cg & 1)) * 16 + ((a_cg >= 2) ? 8 : 0);
  const int b_j = tid & 63, b_kg = tid >> 6;
  const int b_dst = (b_j >> 4) * 512 + ((b_j & 15) + 16 * (b_kg & 1)) * 16 + (b_kg >> 1) * 8;

  v8f acca = {}, accb = {};

  for (int k0 = 0; k0 < NTOK; k0 += 32) {
    *(v8bf*)(As + a_dst) = *(const v8bf*)(A + (size_t)(m0 + a_r) * NTOK + k0 + a_cg * 8);
    *(v8bf*)(Bs + b_dst) = *(const v8bf*)(B + (size_t)(n0 + b_j) * NTOK + k0 + b_kg * 8);
    __syncthreads();
    const int lo = lane * 16;
    v16bf a = *(const v16bf*)(As + mi * 512 + lo);
    v16bf ba = *(const v16bf*)(Bs + njb * 512 + lo);
    v16bf bb = *(const v16bf*)(Bs + (njb + 1) * 512 + lo);
    acca = WMMA_BF16(a, ba, acca);
    accb = WMMA_BF16(a, bb, accb);
    __syncthreads();
  }
  for (int t = 0; t < 2; ++t) {
    v8f p = t ? accb : acca;
    int gn = n0 + (njb + t) * 16 + nl;
    for (int v = 0; v < 8; ++v) {
      int gm = m0 + mi * 16 + v + 8 * half;
      Y[((size_t)gm * NTOK + gn) * CZd + c] = p[v];
    }
  }
}

// ----------------------------- attention core ------------------------------
// 32 query rows (j0..j0+31) vs 256 keys, head dim d (<=32, zero padded).
__device__ void attn_core(__bf16* Qs, __bf16* KVs, float* S,
                          float* red, float* rmax, float* rsum,
                          const float* qb, int qstride,
                          const float* kb, int kstride,
                          const float* vb, int vstride,
                          const float* biasb, int b_rstride, int b_cstride,
                          const float* maskb,
                          float* ob, int ostride, int j0, int d, float scale) {
  const int tid = threadIdx.x, lane = tid & 31, wave = tid >> 5;
  const int half = lane >> 4, nl = lane & 15;
  __bf16* Ps = (__bf16*)S;  // bf16 P fragments alias dead f32 scores

  // Q -> A-fragment layout [rt][lane][e]
  if (tid < 128) {
    const int r = tid >> 2, cg = tid & 3;
    v8bf t;
    for (int i = 0; i < 8; ++i) {
      int cc = cg * 8 + i;
      t[i] = (__bf16)((cc < d) ? qb[(size_t)(j0 + r) * qstride + cc] : 0.0f);
    }
    *(v8bf*)(Qs + (r >> 4) * 512 + ((r & 15) + 16 * (cg & 1)) * 16 + ((cg >= 2) ? 8 : 0)) = t;
  }
  // K -> B-fragment layout [ct(16)][lane][e]  (N = key, K-dim = feature)
  {
    const int r = tid;
    for (int gq = 0; gq < 4; ++gq) {
      v8bf t;
      for (int i = 0; i < 8; ++i) {
        int cc = gq * 8 + i;
        t[i] = (__bf16)((cc < d) ? kb[(size_t)r * kstride + cc] : 0.0f);
      }
      *(v8bf*)(KVs + (r >> 4) * 512 + ((r & 15) + 16 * (gq & 1)) * 16 + (gq >> 1) * 8) = t;
    }
  }
  __syncthreads();

  // S = Q @ K^T (32x256): 32 16x16 tiles, 4 per wave
  for (int t = wave; t < 32; t += 8) {
    int rt = t >> 4, ct = t & 15;
    const int lo = lane * 16;
    v16bf a = *(const v16bf*)(Qs + rt * 512 + lo);
    v16bf b = *(const v16bf*)(KVs + ct * 512 + lo);
    v8f acc = {};
    acc = WMMA_BF16(a, b, acc);
    for (int v = 0; v < 8; ++v) {
      int r = rt * 16 + v + 8 * half;
      int kc = ct * 16 + nl;
      float sv = acc[v] * scale;
      if (biasb) sv += biasb[(size_t)(j0 + r) * b_rstride + (size_t)kc * b_cstride];
      if (maskb) sv += (maskb[kc] - 1.0f) * 1e9f;
      S[r * 256 + kc] = sv;
    }
  }
  __syncthreads();

  // softmax over keys; write P straight into bf16 A-fragment layout
  {
    const int r = tid >> 3, sl = tid & 7;   // thread owns 32 contiguous keys
    const float* Srow = S + r * 256;
    float mx = -1e30f;
#pragma unroll
    for (int i = 0; i < 32; ++i) mx = fmaxf(mx, Srow[sl * 32 + i]);
    red[r * 8 + sl] = mx; __syncthreads();
    if (sl == 0) {
      float m2 = red[r * 8];
      for (int i = 1; i < 8; ++i) m2 = fmaxf(m2, red[r * 8 + i]);
      rmax[r] = m2;
    }
    __syncthreads();
    const float m2 = rmax[r];
    float ev[32]; float sum = 0.f;
#pragma unroll
    for (int i = 0; i < 32; ++i) { ev[i] = __expf(Srow[sl * 32 + i] - m2); sum += ev[i]; }
    __syncthreads();  // all f32 reads of S complete before bf16 overwrite
    const int kt = sl, rt = r >> 4;
#pragma unroll
    for (int g = 0; g < 4; ++g) {
      v8bf t;
#pragma unroll
      for (int i = 0; i < 8; ++i) t[i] = (__bf16)ev[g * 8 + i];
      // kc0 = sl*32 + g*8 ; lane = (r&15) + 16*(g&1) ; e0 = (g>=2)?8:0
      *(v8bf*)(Ps + kt * 1024 + rt * 512 + ((r & 15) + 16 * (g & 1)) * 16 + ((g >= 2) ? 8 : 0)) = t;
    }
    red[r * 8 + sl] = sum; __syncthreads();
    if (sl == 0) {
      float s2 = 0.f;
      for (int i = 0; i < 8; ++i) s2 += red[r * 8 + i];
      rsum[r] = s2;
    }
  }
  __syncthreads();

  // V -> B-fragment layout [kt(8)][cd(2)][lane][e] (K-dim = key, N = feature)
  {
    const int dd = tid & 31, kb0 = (tid >> 5) * 32;
    for (int g = 0; g < 4; ++g) {
      v8bf t;
      for (int i = 0; i < 8; ++i) {
        int krow = kb0 + g * 8 + i;
        t[i] = (__bf16)((dd < d) ? vb[(size_t)krow * vstride + dd] : 0.0f);
      }
      *(v8bf*)(KVs + (tid >> 5) * 1024 + (dd >> 4) * 512 +
               ((dd & 15) + 16 * (g & 1)) * 16 + (g >> 1) * 8) = t;
    }
  }
  __syncthreads();

  // O = P @ V : 4 output tiles on waves 0..3, K loop over 8 key tiles
  if (wave < 4) {
    const int rt = wave >> 1, cd = wave & 1;
    const int lo = lane * 16;
    v8f acc = {};
    for (int kt = 0; kt < 8; ++kt) {
      v16bf a = *(const v16bf*)(Ps + kt * 1024 + rt * 512 + lo);
      v16bf b = *(const v16bf*)(KVs + kt * 1024 + cd * 512 + lo);
      acc = WMMA_BF16(a, b, acc);
    }
    for (int v = 0; v < 8; ++v) {
      int r = rt * 16 + v + 8 * half;
      int cc = cd * 16 + nl;
      if (cc < d)
        ob[(size_t)(j0 + r) * ostride + cc] = acc[v] / rsum[r];
    }
  }
}

__global__ __launch_bounds__(256) void k_tri_attn(const float* q, const float* k,
                                                  const float* v, const float* bias,
                                                  const float* pm, float* o) {
  __shared__ __align__(32) __bf16 Qs[2 * 512];
  __shared__ __align__(32) __bf16 KVs[16 * 512];
  __shared__ __align__(32) float S[32 * 256];
  __shared__ float red[32 * 8];
  __shared__ float rmax[32];
  __shared__ float rsum[32];
  const int i = blockIdx.y >> 2, h = blockIdx.y & 3;
  const size_t base = (size_t)i * NTOK * CZd + h * 32;
  attn_core(Qs, KVs, S, red, rmax, rsum,
            q + base, CZd, k + base, CZd, v + base, CZd,
            bias + h, NTOK * 4, 4, pm + (size_t)i * NTOK,
            o + base, CZd, blockIdx.x * 32, 32, 0.17677669529663687f);
}

__global__ __launch_bounds__(256) void k_apb_attn(const float* q, const float* k,
                                                  const float* v, const float* zb,
                                                  const float* sm, float* o) {
  __shared__ __align__(32) __bf16 Qs[2 * 512];
  __shared__ __align__(32) __bf16 KVs[16 * 512];
  __shared__ __align__(32) float S[32 * 256];
  __shared__ float red[32 * 8];
  __shared__ float rmax[32];
  __shared__ float rsum[32];
  const int h = blockIdx.y;
  attn_core(Qs, KVs, S, red, rmax, rsum,
            q + h * 24, 384, k + h * 24, 384, v + h * 24, 384,
            zb + h, NTOK * 16, 16, sm,
            o + h * 24, 384, blockIdx.x * 32, 24, 0.20412414523193154f);
}

// ------------------------------ elementwise --------------------------------
__global__ __launch_bounds__(256) void k_mul(const float* __restrict__ a,
                                             const float* __restrict__ b,
                                             float* __restrict__ y, size_t n4) {
  size_t i = (size_t)blockIdx.x * 256 + threadIdx.x;
  if (i < n4) {
    float4 av = ((const float4*)a)[i], bv = ((const float4*)b)[i], o;
    o.x = av.x * bv.x; o.y = av.y * bv.y; o.z = av.z * bv.z; o.w = av.w * bv.w;
    ((float4*)y)[i] = o;
  }
}

// dst[c][x][y] = src[(sw? y*256+x : x*256+y)*128 + c]; src-contiguous reads
__global__ __launch_bounds__(256) void k_pack(const float* __restrict__ src,
                                              __bf16* __restrict__ dst, int sw) {
  size_t t = (size_t)blockIdx.x * 256 + threadIdx.x;  // 65536 rows * 16 cgroups
  int row = (int)(t >> 4), c0 = (int)(t & 15) * 8;
  int x = sw ? (row & 255) : (row >> 8);
  int y = sw ? (row >> 8) : (row & 255);
  float4 f0 = *(const float4*)(src + (size_t)row * CZd + c0);
  float4 f1 = *(const float4*)(src + (size_t)row * CZd + c0 + 4);
  size_t dbase = (size_t)x * NTOK + y;
  dst[(size_t)(c0 + 0) * 65536 + dbase] = (__bf16)f0.x;
  dst[(size_t)(c0 + 1) * 65536 + dbase] = (__bf16)f0.y;
  dst[(size_t)(c0 + 2) * 65536 + dbase] = (__bf16)f0.z;
  dst[(size_t)(c0 + 3) * 65536 + dbase] = (__bf16)f0.w;
  dst[(size_t)(c0 + 4) * 65536 + dbase] = (__bf16)f1.x;
  dst[(size_t)(c0 + 5) * 65536 + dbase] = (__bf16)f1.y;
  dst[(size_t)(c0 + 6) * 65536 + dbase] = (__bf16)f1.z;
  dst[(size_t)(c0 + 7) * 65536 + dbase] = (__bf16)f1.w;
}

// dst[(j*256+i)*C + c] = src[(i*256+j)*C + c]
__global__ __launch_bounds__(256) void k_transpose_ij(const float* __restrict__ src,
                                                      float* __restrict__ dst, int C) {
  if (C == 1) {
    size_t idx = (size_t)blockIdx.x * 256 + threadIdx.x;
    int i = (int)(idx >> 8), j = (int)(idx & 255);
    dst[(size_t)j * NTOK + i] = src[idx];
    return;
  }
  size_t t = (size_t)blockIdx.x * 256 + threadIdx.x;  // rows * (C/4)
  const int Cq = C >> 2;
  int row = (int)(t / Cq), c4 = (int)(t % Cq);
  int i = row >> 8, j = row & 255;
  ((float4*)(dst + ((size_t)j * NTOK + i) * C))[c4] =
      ((const float4*)(src + (size_t)row * C))[c4];
}

// z[(j*256+i)*C+c] += t[(i*256+j)*C+c]
__global__ __launch_bounds__(256) void k_add_transpose(float* z, const float* __restrict__ t,
                                                       int C) {
  size_t g = (size_t)blockIdx.x * 256 + threadIdx.x;  // rows * (C/4)
  const int Cq = C >> 2;
  int row = (int)(g / Cq), c4 = (int)(g % Cq);
  int i = row >> 8, j = row & 255;
  float4 tv = ((const float4*)(t + (size_t)row * C))[c4];
  float4* zp = (float4*)(z + ((size_t)j * NTOK + i) * C) + c4;
  float4 zv = *zp;
  zv.x += tv.x; zv.y += tv.y; zv.z += tv.z; zv.w += tv.w;
  *zp = zv;
}

// ------------------------------- host side ---------------------------------
// params flattened in jax sorted-key order after s, z, single_mask, pair_mask
enum {
  P_APB_BQ = 4, P_APB_LN_B, P_APB_LN_G, P_APB_WB, P_APB_WG, P_APB_WK, P_APB_WO,
  P_APB_WQ, P_APB_WV, P_APB_ZLN_B, P_APB_ZLN_G,
  P_PT_LN_B, P_PT_LN_G, P_PT_WA, P_PT_WB, P_PT_WO,
  P_ST_LN_B, P_ST_LN_G, P_ST_WA, P_ST_WB, P_ST_WO,
  P_TAE_LN_B, P_TAE_LN_G, P_TAE_WB, P_TAE_WG, P_TAE_WK, P_TAE_WO, P_TAE_WQ, P_TAE_WV,
  P_TAS_LN_B, P_TAS_LN_G, P_TAS_WB, P_TAS_WG, P_TAS_WK, P_TAS_WO, P_TAS_WQ, P_TAS_WV,
  P_TMI_LN_B, P_TMI_LN_G, P_TMI_LNO_B, P_TMI_LNO_G, P_TMI_WG, P_TMI_WGA,
  P_TMI_WGB, P_TMI_WO, P_TMI_WPA, P_TMI_WPB,
  P_TMO_LN_B, P_TMO_LN_G, P_TMO_LNO_B, P_TMO_LNO_G, P_TMO_WG, P_TMO_WGA,
  P_TMO_WGB, P_TMO_WO, P_TMO_WPA, P_TMO_WPB
};

extern "C" void kernel_launch(void* const* d_in, const int* in_sizes, int n_in,
                              void* d_out, int out_size, void* d_ws, size_t ws_size,
                              hipStream_t stream) {
  (void)n_in; (void)out_size; (void)ws_size;
  const float* sm = (const float*)d_in[2];
  const float* pm = (const float*)d_in[3];
  float* s = (float*)d_out;
  float* z = s + NTOK * 384;
  hipMemcpyAsync(s, d_in[0], sizeof(float) * NTOK * 384, hipMemcpyDeviceToDevice, stream);
  hipMemcpyAsync(z, d_in[1], sizeof(float) * PAIR_ELEMS, hipMemcpyDeviceToDevice, stream);

  const size_t B = PAIR_ELEMS;
  float* W0 = (float*)d_ws;
  float* W1 = W0 + B;  float* W2 = W1 + B;  float* W3 = W2 + B;
  float* W4 = W3 + B;  float* W5 = W4 + B;
  __bf16* At = (__bf16*)(W5 + B);
  __bf16* Bt = At + B;
  float* SMALL = (float*)(Bt + B);
  float* zb = SMALL;              // 65536*16
  float* tbias = zb + 1048576;    // 65536*4
  float* pmt = tbias + 262144;    // 65536
  float* hs = pmt + 65536;        // 256*384
  float* aq = hs + 98304;  float* ak = aq + 98304;  float* av = ak + 98304;
  float* ag = av + 98304;  float* ao = ag + 98304;
  float* stp = ao + 98304;        // 256*1536

  auto par = [&](int idx, int l) -> const float* {
    return (const float*)d_in[idx] + (size_t)(in_sizes[idx] / 2) * l;
  };
  auto gemm = [&](const float* A1, const float* Wm1, const float* A2, const float* Wm2,
                  const float* cb, const float* rs, const float* res, float* Y,
                  int M, int N, int K, int act) {
    dim3 g((N + 63) / 64, M / 64);
    k_gemm<<<g, 256, 0, stream>>>(A1, Wm1, A2, Wm2, cb, rs, res, Y, M, N, K, act);
  };
  const unsigned PACK_GRID = 65536 * 16 / 256;       // 4096
  const unsigned TR_GRID   = 65536 * 32 / 256;       // 8192 (C=128, float4)

  k_transpose_ij<<<65536 / 256, 256, 0, stream>>>(pm, pmt, 1);

  for (int l = 0; l < 2; ++l) {
    // ---------------- triangle multiplication (outgoing, incoming) --------
    for (int pass = 0; pass < 2; ++pass) {
      const bool outg = (pass == 0);
      const int b0 = outg ? P_TMO_LN_B : P_TMI_LN_B;  // +0 ln_b,+1 ln_g,+2 lno_b,+3 lno_g,+4 wg,+5 wga,+6 wgb,+7 wo,+8 wpa,+9 wpb
      const int sw = outg ? 0 : 1;
      k_ln<<<65536, 128, 0, stream>>>(z, par(b0 + 1, l), par(b0 + 0, l), W0, 128);
      gemm(W0, par(b0 + 5, l), W0, par(b0 + 8, l), nullptr, pm, nullptr, W1, 65536, 128, 128, 1);
      gemm(W0, par(b0 + 6, l), W0, par(b0 + 9, l), nullptr, pm, nullptr, W2, 65536, 128, 128, 1);
      k_pack<<<PACK_GRID, 256, 0, stream>>>(W1, At, sw);  // A: [c][i][k]
      k_pack<<<PACK_GRID, 256, 0, stream>>>(W2, Bt, sw);  // B: [c][j][k]
      k_tri_gemm<<<dim3(4, 4, 128), 256, 0, stream>>>(At, Bt, W3);
      k_ln<<<65536, 128, 0, stream>>>(W3, par(b0 + 3, l), par(b0 + 2, l), W4, 128);
      gemm(W0, par(b0 + 4, l), W4, par(b0 + 7, l), nullptr, nullptr, z, z, 65536, 128, 128, 1);
    }
    // ---------------- triangle attention (starting, ending) ---------------
    for (int pass = 0; pass < 2; ++pass) {
      const bool start = (pass == 0);
      const int b0 = start ? P_TAS_LN_B : P_TAE_LN_B;  // +0 ln_b,+1 ln_g,+2 wb,+3 wg,+4 wk,+5 wo,+6 wq,+7 wv
      const float* zsrc = z; const float* msk = pm;
      if (!start) {
        k_transpose_ij<<<TR_GRID, 256, 0, stream>>>(z, W5, 128);
        zsrc = W5; msk = pmt;
      }
      k_ln<<<65536, 128, 0, stream>>>(zsrc, par(b0 + 1, l), par(b0 + 0, l), W0, 128);
      gemm(W0, par(b0 + 6, l), nullptr, nullptr, nullptr, nullptr, nullptr, W1, 65536, 128, 128, 0);
      gemm(W0, par(b0 + 4, l), nullptr, nullptr, nullptr, nullptr, nullptr, W2, 65536, 128, 128, 0);
      gemm(W0, par(b0 + 7, l), nullptr, nullptr, nullptr, nullptr, nullptr, W3, 65536, 128, 128, 0);
      gemm(W0, par(b0 + 3, l), nullptr, nullptr, nullptr, nullptr, nullptr, W4, 65536, 128, 128, 3);
      gemm(W0, par(b0 + 2, l), nullptr, nullptr, nullptr, nullptr, nullptr, tbias, 65536, 4, 128, 0);
      k_tri_attn<<<dim3(8, 1024), 256, 0, stream>>>(W1, W2, W3, tbias, msk, W5);
      k_mul<<<(unsigned)(B / 1024), 256, 0, stream>>>(W4, W5, W4, B / 4);
      if (start) {
        gemm(W4, par(b0 + 5, l), nullptr, nullptr, nullptr, nullptr, z, z, 65536, 128, 128, 0);
      } else {
        gemm(W4, par(b0 + 5, l), nullptr, nullptr, nullptr, nullptr, nullptr, W0, 65536, 128, 128, 0);
        k_add_transpose<<<TR_GRID, 256, 0, stream>>>(z, W0, 128);
      }
    }
    // ---------------- pair SwiGLU transition (chunked over rows) ----------
    k_ln<<<65536, 128, 0, stream>>>(z, par(P_PT_LN_G, l), par(P_PT_LN_B, l), W0, 128);
    for (int ch = 0; ch < 4; ++ch) {
      const int r0 = ch * 16384;
      gemm(W0 + (size_t)r0 * 128, par(P_PT_WA, l), W0 + (size_t)r0 * 128, par(P_PT_WB, l),
           nullptr, nullptr, nullptr, W1, 16384, 512, 128, 2);
      gemm(W1, par(P_PT_WO, l), nullptr, nullptr, nullptr, pm + r0,
           z + (size_t)r0 * 128, z + (size_t)r0 * 128, 16384, 128, 512, 0);
    }
    // ---------------- attention-pair-bias on single -----------------------
    k_ln<<<65536, 128, 0, stream>>>(z, par(P_APB_ZLN_G, l), par(P_APB_ZLN_B, l), W0, 128);
    gemm(W0, par(P_APB_WB, l), nullptr, nullptr, nullptr, nullptr, nullptr, zb, 65536, 16, 128, 0);
    k_ln<<<256, 128, 0, stream>>>(s, par(P_APB_LN_G, l), par(P_APB_LN_B, l), hs, 384);
    gemm(hs, par(P_APB_WQ, l), nullptr, nullptr, par(P_APB_BQ, l), nullptr, nullptr, aq, 256, 384, 384, 0);
    gemm(hs, par(P_APB_WK, l), nullptr, nullptr, nullptr, nullptr, nullptr, ak, 256, 384, 384, 0);
    gemm(hs, par(P_APB_WV, l), nullptr, nullptr, nullptr, nullptr, nullptr, av, 256, 384, 384, 0);
    gemm(hs, par(P_APB_WG, l), nullptr, nullptr, nullptr, nullptr, nullptr, ag, 256, 384, 384, 3);
    k_apb_attn<<<dim3(8, 16), 256, 0, stream>>>(aq, ak, av, zb, sm, ao);
    k_mul<<<(256 * 384) / 1024, 256, 0, stream>>>(ag, ao, ag, (size_t)256 * 384 / 4);
    gemm(ag, par(P_APB_WO, l), nullptr, nullptr, nullptr, nullptr, s, s, 256, 384, 384, 0);
    // ---------------- single SwiGLU transition ----------------------------
    k_ln<<<256, 128, 0, stream>>>(s, par(P_ST_LN_G, l), par(P_ST_LN_B, l), hs, 384);
    gemm(hs, par(P_ST_WA, l), hs, par(P_ST_WB, l), nullptr, nullptr, nullptr, stp, 256, 1536, 384, 2);
    gemm(stp, par(P_ST_WO, l), nullptr, nullptr, nullptr, sm, s, s, 256, 384, 1536, 0);
  }
}
